// SSMPolicy_67353677136131
// MI455X (gfx1250) — matrix operand order repaired and checked by reading, not compiled, not run on hardware
//
#include <hip/hip_runtime.h>
#include <hip/hip_bf16.h>
#include <cstdint>

#define BB   128
#define TT   256
#define SS   512
#define HH   1024
#define AA   64
#define EPSL 1e-5f

typedef __attribute__((ext_vector_type(16))) _Float16     v16h;
typedef __attribute__((ext_vector_type(8)))  _Float16     v8h;
typedef __attribute__((ext_vector_type(8)))  float        v8f;
typedef __attribute__((ext_vector_type(4)))  unsigned int v4u;
typedef __attribute__((ext_vector_type(8)))  unsigned int v8u;

__device__ __forceinline__ v8f zero8() {
  v8f z;
#pragma unroll
  for (int i = 0; i < 8; ++i) z[i] = 0.0f;
  return z;
}

// A-fragment per ISA 16-bit A 16x32 layout:
// elements 0..7  -> K = hf*8 + e        (VGPR0..3)
// elements 8..15 -> K = 16 + hf*8 + e-8 (VGPR4..7)
__device__ __forceinline__ v16h load_a16(const _Float16* s, int ln, int hf,
                                         int kbase, int stride) {
  const _Float16* p = s + ln * stride + kbase + hf * 8;
  v8h lo = *(const v8h*)(p);
  v8h hi = *(const v8h*)(p + 16);
  v16h a;
#pragma unroll
  for (int i = 0; i < 8; ++i) { a[i] = lo[i]; a[8 + i] = hi[i]; }
  return a;
}

__device__ __forceinline__ float sigm(float x) { return 1.0f / (1.0f + __expf(-x)); }

// issue one per-lane async 16B global->LDS copy (ASYNCcnt path)
__device__ __forceinline__ void async_cp16(const _Float16* g, const _Float16* lds) {
  unsigned int la = (unsigned int)(uintptr_t)lds;  // LDS aperture: addr[31:0] is LDS offset
  asm volatile("global_load_async_to_lds_b128 %0, %1, off"
               :: "v"(la), "v"(g) : "memory");
}
__device__ __forceinline__ void wait_async0() {
  asm volatile("s_wait_asynccnt 0x0" ::: "memory");
}

// ---------------- weight pack: f32 [K,N(ld)] -> f16 WMMA B-layout ----------------
// out[((kt*NT+nt)*32 + lane)*16 + i] = W[kt*32 + (i>>3)*16 + (lane>>4)*8 + (i&7)][nt*16 + (lane&15)]
__global__ void pack_w(const float* __restrict__ W, _Float16* __restrict__ out,
                       int K, int N, int ld) {
  int idx = blockIdx.x * blockDim.x + threadIdx.x;
  if (idx >= K * N) return;
  int i    = idx & 15;
  int lane = (idx >> 4) & 31;
  int tile = idx >> 9;
  int nT   = N >> 4;
  int nt   = tile % nT;
  int kt   = tile / nT;
  int k = kt * 32 + ((i >> 3) << 4) + ((lane >> 4) << 3) + (i & 7);
  int n = nt * 16 + (lane & 15);
  out[idx] = (_Float16)W[(size_t)k * ld + n];
}

// ---------------- encoder: x = silu(LN(obs@We + be)) ----------------
__global__ void __launch_bounds__(512) k_encode(
    const float* __restrict__ obs, const _Float16* __restrict__ pWe,
    const float* __restrict__ be, const float* __restrict__ ge,
    const float* __restrict__ beln, _Float16* __restrict__ xOut) {
  __shared__ _Float16 As[16 * SS];
  __shared__ float rs[16], rq[16];
  const int tid  = threadIdx.x;
  const int row0 = blockIdx.x * 16;
#pragma unroll 4
  for (int idx = tid; idx < 16 * SS; idx += 512) {
    int r = idx >> 9, c = idx & (SS - 1);
    As[idx] = (_Float16)obs[(size_t)(row0 + r) * SS + c];
  }
  if (tid < 16) { rs[tid] = 0.0f; rq[tid] = 0.0f; }
  __syncthreads();

  const int lane = tid & 31, wave = tid >> 5;
  const int ln = lane & 15, hf = lane >> 4;
  v8f acc[4];
#pragma unroll
  for (int j = 0; j < 4; ++j) acc[j] = zero8();

  const int NT = HH / 16;  // 64 n-tiles
  for (int kt = 0; kt < SS / 32; ++kt) {
    v16h a = load_a16(As, ln, hf, kt * 32, SS);
    if (kt + 1 < SS / 32)
      __builtin_prefetch(pWe + ((size_t)(((kt + 1) * NT + wave * 4) * 32 + lane) << 4), 0, 0);
#pragma unroll
    for (int j = 0; j < 4; ++j) {
      int nt = wave * 4 + j;
      v16h b = *(const v16h*)(pWe + ((size_t)((kt * NT + nt) * 32 + lane) << 4));
      acc[j] = __builtin_amdgcn_wmma_f32_16x16x32_f16(false, a, false, b,
                                                      (short)0, acc[j], false, false);
    }
  }
  // epilogue: +bias, LN row reduction, SiLU
  float psum[8], psq[8];
#pragma unroll
  for (int r = 0; r < 8; ++r) { psum[r] = 0.0f; psq[r] = 0.0f; }
#pragma unroll
  for (int j = 0; j < 4; ++j) {
    int col = (wave * 4 + j) * 16 + ln;
    float bb = be[col];
#pragma unroll
    for (int r = 0; r < 8; ++r) {
      float v = acc[j][r] + bb;
      acc[j][r] = v;
      psum[r] += v; psq[r] += v * v;
    }
  }
#pragma unroll
  for (int r = 0; r < 8; ++r) {
    atomicAdd(&rs[r + 8 * hf], psum[r]);
    atomicAdd(&rq[r + 8 * hf], psq[r]);
  }
  __syncthreads();
#pragma unroll
  for (int j = 0; j < 4; ++j) {
    int col = (wave * 4 + j) * 16 + ln;
    float g = ge[col], bl = beln[col];
#pragma unroll
    for (int r = 0; r < 8; ++r) {
      int row = r + 8 * hf;
      float mu  = rs[row] * (1.0f / HH);
      float var = rq[row] * (1.0f / HH) - mu * mu;
      float y = (acc[j][r] - mu) * rsqrtf(var + EPSL) * g + bl;
      float s = y * sigm(y);
      xOut[(size_t)(row0 + row) * HH + col] = (_Float16)s;
    }
  }
}

// ---------------- in_proj: u = tanh(x@Win[:, :H] + bin[:H]) ----------------
__global__ void __launch_bounds__(512) k_inproj(
    const _Float16* __restrict__ x, const _Float16* __restrict__ pWin,
    const float* __restrict__ binp, _Float16* __restrict__ uOut) {
  __shared__ _Float16 As[16 * HH];
  const int tid = threadIdx.x;
  const int row0 = blockIdx.x * 16;
  // async f16 staging: 2048 16-byte chunks, 4 per thread
  for (int idx = tid; idx < (16 * HH) / 8; idx += 512) {
    int r = idx >> 7;           // 128 chunks per row
    int c = (idx & 127) << 3;
    async_cp16(x + (size_t)(row0 + r) * HH + c, As + (size_t)r * HH + c);
  }
  wait_async0();
  __syncthreads();

  const int lane = tid & 31, wave = tid >> 5;
  const int ln = lane & 15, hf = lane >> 4;
  v8f acc[4];
#pragma unroll
  for (int j = 0; j < 4; ++j) acc[j] = zero8();

  const int NT = HH / 16;
  for (int kt = 0; kt < HH / 32; ++kt) {
    v16h a = load_a16(As, ln, hf, kt * 32, HH);
#pragma unroll
    for (int j = 0; j < 4; ++j) {
      int nt = wave * 4 + j;
      v16h b = *(const v16h*)(pWin + ((size_t)((kt * NT + nt) * 32 + lane) << 4));
      acc[j] = __builtin_amdgcn_wmma_f32_16x16x32_f16(false, a, false, b,
                                                      (short)0, acc[j], false, false);
    }
  }
#pragma unroll
  for (int j = 0; j < 4; ++j) {
    int col = (wave * 4 + j) * 16 + ln;
    float bb = binp[col];
#pragma unroll
    for (int r = 0; r < 8; ++r) {
      int row = r + 8 * hf;
      uOut[(size_t)(row0 + row) * HH + col] = (_Float16)tanhf(acc[j][r] + bb);
    }
  }
}

// ---------------- init h ----------------
__global__ void k_init_h(const float* __restrict__ hidden,
                         float* __restrict__ hF32, _Float16* __restrict__ hF16) {
  int i = blockIdx.x * blockDim.x + threadIdx.x;
  if (i < BB * HH) { float v = hidden[i]; hF32[i] = v; hF16[i] = (_Float16)v; }
}

// ---------------- one recurrence step ----------------
// s = tanh(h@Ws + bs); g = sigmoid(h@Wg + bg); h = LN(h + g*(u_t + s))
__global__ void __launch_bounds__(512) k_step(
    const _Float16* __restrict__ pWs, const _Float16* __restrict__ pWg,
    const float* __restrict__ bs, const float* __restrict__ bg,
    const float* __restrict__ gn, const float* __restrict__ bn,
    const _Float16* __restrict__ u,             // [B*T, H], row = b*T + t
    float* __restrict__ hF32, _Float16* __restrict__ hF16,
    _Float16* __restrict__ hs,                  // [T*B, H]
    float* __restrict__ hlast, int t) {
  __shared__ _Float16 As[16 * HH];
  __shared__ float rs[16], rq[16];
  const int tid = threadIdx.x;
  const int b0 = blockIdx.x * 16;
  const int lane = tid & 31, wave = tid >> 5;
  const int ln = lane & 15, hf = lane >> 4;

  // ---- TDM: stage the 16xH f16 h-tile into LDS with one tensor_load_to_lds ----
  if (wave == 0) {
    uint64_t ga = (uint64_t)(uintptr_t)(hF16 + (size_t)b0 * HH);  // tile start
    unsigned int la = (unsigned int)(uintptr_t)(&As[0]);          // LDS byte offset
    // D# group 0: count=1 | lds_addr | global_addr[56:0] | type=2
    v4u g0 = { 1u, la, (unsigned int)ga,
               (unsigned int)((ga >> 32) & 0x01FFFFFFu) | (2u << 30) };
    // D# group 1: 2D tensor, data_size=2B, dim0=HH, dim1=16, tile=HHx16, stride0=HH
    v8u g1 = { (1u << 16),                               // data_size=1 (2 bytes)
               ((unsigned int)(HH & 0xFFFF)) << 16,      // tensor_dim0[15:0]
               ((unsigned int)(HH >> 16)) | (16u << 16), // dim0[31:16] | tensor_dim1[15:0]
               ((unsigned int)HH) << 16,                 // dim1[31:16]=0 | tile_dim0
               16u,                                      // tile_dim1=16, tile_dim2=0
               (unsigned int)HH,                         // tensor_dim0_stride[31:0]
               0u, 0u };
    asm volatile("tensor_load_to_lds %0, %1" :: "s"(g0), "s"(g1) : "memory");
    asm volatile("s_wait_tensorcnt 0x0" ::: "memory");
  }
  if (tid < 16) { rs[tid] = 0.0f; rq[tid] = 0.0f; }
  __syncthreads();

  v8f accS[4], accG[4];
#pragma unroll
  for (int j = 0; j < 4; ++j) { accS[j] = zero8(); accG[j] = zero8(); }

  const int NT = HH / 16;
  for (int kt = 0; kt < HH / 32; ++kt) {
    v16h a = load_a16(As, ln, hf, kt * 32, HH);
    if (kt + 1 < HH / 32)
      __builtin_prefetch(pWs + ((size_t)(((kt + 1) * NT + wave * 4) * 32 + lane) << 4), 0, 0);
#pragma unroll
    for (int j = 0; j < 4; ++j) {
      int nt = wave * 4 + j;
      size_t boff = ((size_t)((kt * NT + nt) * 32 + lane) << 4);
      v16h bS = *(const v16h*)(pWs + boff);
      accS[j] = __builtin_amdgcn_wmma_f32_16x16x32_f16(false, a, false, bS,
                                                       (short)0, accS[j], false, false);
      v16h bG = *(const v16h*)(pWg + boff);
      accG[j] = __builtin_amdgcn_wmma_f32_16x16x32_f16(false, a, false, bG,
                                                       (short)0, accG[j], false, false);
    }
  }
  // v = h + g*(u+s) ; then LN over row
  float psum[8], psq[8];
#pragma unroll
  for (int r = 0; r < 8; ++r) { psum[r] = 0.0f; psq[r] = 0.0f; }
#pragma unroll
  for (int j = 0; j < 4; ++j) {
    int col = (wave * 4 + j) * 16 + ln;
    float bsv = bs[col], bgv = bg[col];
#pragma unroll
    for (int r = 0; r < 8; ++r) {
      int bb = b0 + r + 8 * hf;
      float sv = tanhf(accS[j][r] + bsv);
      float gv = sigm(accG[j][r] + bgv);
      float uv = (float)u[((size_t)bb * TT + t) * HH + col];
      float hv = hF32[(size_t)bb * HH + col];
      float v  = hv + gv * (uv + sv);
      accS[j][r] = v;
      psum[r] += v; psq[r] += v * v;
    }
  }
#pragma unroll
  for (int r = 0; r < 8; ++r) {
    atomicAdd(&rs[r + 8 * hf], psum[r]);
    atomicAdd(&rq[r + 8 * hf], psq[r]);
  }
  __syncthreads();
#pragma unroll
  for (int j = 0; j < 4; ++j) {
    int col = (wave * 4 + j) * 16 + ln;
    float g = gn[col], b = bn[col];
#pragma unroll
    for (int r = 0; r < 8; ++r) {
      int row = r + 8 * hf;
      int bb = b0 + row;
      float mu  = rs[row] * (1.0f / HH);
      float var = rq[row] * (1.0f / HH) - mu * mu;
      float hn = (accS[j][r] - mu) * rsqrtf(var + EPSL) * g + b;
      hF32[(size_t)bb * HH + col] = hn;
      hF16[(size_t)bb * HH + col] = (_Float16)hn;
      hs[((size_t)t * BB + bb) * HH + col] = (_Float16)hn;
      if (t == TT - 1) hlast[(size_t)bb * HH + col] = hn;
    }
  }
}

// ---------------- policy head: logits = hs@Wp + bp ----------------
__global__ void __launch_bounds__(128) k_policy(
    const _Float16* __restrict__ hs, const _Float16* __restrict__ pWp,
    const float* __restrict__ bp, float* __restrict__ out) {
  __shared__ _Float16 As[16 * HH];
  const int tid = threadIdx.x;
  const int r0 = blockIdx.x * 16;  // row = t*B + b
  for (int idx = tid; idx < (16 * HH) / 8; idx += 128) {
    int r = idx >> 7;
    int c = (idx & 127) << 3;
    async_cp16(hs + (size_t)(r0 + r) * HH + c, As + (size_t)r * HH + c);
  }
  wait_async0();
  __syncthreads();

  const int lane = tid & 31, wave = tid >> 5;  // 4 waves, 1 n-tile each
  const int ln = lane & 15, hf = lane >> 4;
  v8f acc = zero8();
  const int NT = AA / 16;  // 4
  for (int kt = 0; kt < HH / 32; ++kt) {
    v16h a = load_a16(As, ln, hf, kt * 32, HH);
    v16h b = *(const v16h*)(pWp + ((size_t)((kt * NT + wave) * 32 + lane) << 4));
    acc = __builtin_amdgcn_wmma_f32_16x16x32_f16(false, a, false, b,
                                                 (short)0, acc, false, false);
  }
  int col = wave * 16 + ln;
  float bias = bp[col];
#pragma unroll
  for (int r = 0; r < 8; ++r) {
    int rg = r0 + r + 8 * hf;
    int tt = rg >> 7;        // / B(=128)
    int b  = rg & (BB - 1);  // % B
    out[((size_t)b * TT + tt) * AA + col] = acc[r] + bias;
  }
}

extern "C" void kernel_launch(void* const* d_in, const int* in_sizes, int n_in,
                              void* d_out, int out_size, void* d_ws, size_t ws_size,
                              hipStream_t stream) {
  const float* obs    = (const float*)d_in[0];
  const float* hidden = (const float*)d_in[1];
  const float* We   = (const float*)d_in[2];
  const float* be   = (const float*)d_in[3];
  const float* ge   = (const float*)d_in[4];
  const float* beln = (const float*)d_in[5];
  const float* Win  = (const float*)d_in[6];
  const float* binp = (const float*)d_in[7];
  const float* Ws   = (const float*)d_in[8];
  const float* bs   = (const float*)d_in[9];
  const float* Wg   = (const float*)d_in[10];
  const float* bg   = (const float*)d_in[11];
  const float* gn   = (const float*)d_in[12];
  const float* bn   = (const float*)d_in[13];
  const float* Wp   = (const float*)d_in[14];
  const float* bp   = (const float*)d_in[15];

  float* logits = (float*)d_out;                       // [B,T,A]
  float* hlast  = logits + (size_t)BB * TT * AA;       // [B,H]

  char* ws = (char*)d_ws;
  size_t off = 0;
  auto alloc = [&](size_t bytes) -> void* {
    void* p = ws + off;
    off += (bytes + 255) & ~(size_t)255;
    return p;
  };
  _Float16* pWe  = (_Float16*)alloc((size_t)SS * HH * 2);
  _Float16* pWin = (_Float16*)alloc((size_t)HH * HH * 2);
  _Float16* pWs  = (_Float16*)alloc((size_t)HH * HH * 2);
  _Float16* pWg  = (_Float16*)alloc((size_t)HH * HH * 2);
  _Float16* pWp  = (_Float16*)alloc((size_t)HH * AA * 2);
  _Float16* pX   = (_Float16*)alloc((size_t)BB * TT * HH * 2);
  _Float16* pU   = (_Float16*)alloc((size_t)BB * TT * HH * 2);
  _Float16* pHs  = (_Float16*)alloc((size_t)TT * BB * HH * 2);
  float*    hF32 = (float*)   alloc((size_t)BB * HH * 4);
  _Float16* hF16 = (_Float16*)alloc((size_t)BB * HH * 2);

  // 1) pack weights to f16 WMMA B-layout
  pack_w<<<(SS * HH + 255) / 256, 256, 0, stream>>>(We,  pWe,  SS, HH, HH);
  pack_w<<<(HH * HH + 255) / 256, 256, 0, stream>>>(Win, pWin, HH, HH, 2 * HH); // first H cols
  pack_w<<<(HH * HH + 255) / 256, 256, 0, stream>>>(Ws,  pWs,  HH, HH, HH);
  pack_w<<<(HH * HH + 255) / 256, 256, 0, stream>>>(Wg,  pWg,  HH, HH, HH);
  pack_w<<<(HH * AA + 255) / 256, 256, 0, stream>>>(Wp,  pWp,  HH, AA, AA);

  // 2) encoder + in_proj (batched over B*T rows, 16 rows per block)
  k_encode<<<(BB * TT) / 16, 512, 0, stream>>>(obs, pWe, be, ge, beln, pX);
  k_inproj<<<(BB * TT) / 16, 512, 0, stream>>>(pX, pWin, binp, pU);

  // 3) recurrence: 256 serial steps, 8 blocks of 16 batch-rows each
  k_init_h<<<(BB * HH + 255) / 256, 256, 0, stream>>>(hidden, hF32, hF16);
  for (int t = 0; t < TT; ++t) {
    k_step<<<BB / 16, 512, 0, stream>>>(pWs, pWg, bs, bg, gn, bn, pU,
                                        hF32, hF16, pHs, hlast, t);
  }

  // 4) policy head
  k_policy<<<(TT * BB) / 16, 128, 0, stream>>>(pHs, pWp, bp, logits);
}
